// MyLSTM_42331197670183
// MI455X (gfx1250) — matrix operand order repaired and checked by reading, not compiled, not run on hardware
//
#include <hip/hip_runtime.h>

// ---------------------------------------------------------------------------
// MyLSTM for MI455X (gfx1250):
//   K1 prep_wz : repack 4 gate weight matrices -> f16 Wz[160][512] (+bias bz[512])
//   K2 embed   : e = sigmoid(x @ W_emb + b)  as WMMA f16 GEMM, stored f16 [T][64][32]
//   K3 lstm    : single persistent 1024-thread workgroup (32 waves, one WGP),
//                weights resident in VGPRs, h/c in LDS, per-step e_t pulled via
//                global_load_async_to_lds_b32, gates via v_wmma_f32_16x16x32_f16
// ---------------------------------------------------------------------------

typedef _Float16 f16;
typedef __attribute__((ext_vector_type(16))) _Float16 v16h;
typedef __attribute__((ext_vector_type(8)))  _Float16 v8h;
typedef __attribute__((ext_vector_type(8)))  float    v8f;

#define B_SZ   64
#define T_SZ   4096
#define D_IN   64
#define D_EMB  24
#define H_SZ   128
#define KZ     160          // padded K: 32 (e, 24 real) + 128 (h)
#define NG     512          // 4 gates * 128

__device__ __forceinline__ float sigmoidf_(float v) {
    return __builtin_amdgcn_rcpf(1.0f + __expf(-v));
}
__device__ __forceinline__ float tanhf_(float v) {
    // tanh(v) = 1 - 2/(1+exp(2v)); saturates correctly at +-1 for |v| large
    return 1.0f - 2.0f * __builtin_amdgcn_rcpf(1.0f + __expf(2.0f * v));
}

// ---------------------------------------------------------------------------
// K1: build Wz f16 [160][512] (row k, col = gate*128 + j) and bz f32 [512]
//     z layout is [e(0..23) | zero(24..31) | h(32..159)], so
//     Wz rows:  k<24 -> W_g[k], 24<=k<32 -> 0, k>=32 -> W_g[k-8]
// ---------------------------------------------------------------------------
__global__ void prep_wz_kernel(const float* __restrict__ Wf, const float* __restrict__ bf,
                               const float* __restrict__ Wi, const float* __restrict__ bi,
                               const float* __restrict__ Wc, const float* __restrict__ bc,
                               const float* __restrict__ Wo, const float* __restrict__ bo,
                               f16* __restrict__ Wz, float* __restrict__ bz)
{
    int gid = blockIdx.x * 256 + threadIdx.x;
    if (gid < NG) {
        int gate = gid >> 7, j = gid & 127;
        const float* bs = (gate == 0) ? bf : (gate == 1) ? bi : (gate == 2) ? bc : bo;
        bz[gid] = bs[j];
    }
    if (gid < KZ * NG) {
        int k = gid >> 9, col = gid & (NG - 1);
        int gate = col >> 7, j = col & 127;
        const float* Ws = (gate == 0) ? Wf : (gate == 1) ? Wi : (gate == 2) ? Wc : Wo;
        float v = 0.0f;
        if (k < 24)       v = Ws[k * H_SZ + j];
        else if (k >= 32) v = Ws[(k - 8) * H_SZ + j];
        Wz[gid] = (f16)v;
    }
}

// ---------------------------------------------------------------------------
// K2: embedding GEMM. x flat rows r = b*T + t (x is [B,T,64]); output stored
//     time-major padded: ep[t][b][0..31] f16 (cols >=24 forced to zero).
//     256 threads = 8 waves/WG, wave owns one 16-row M-tile, N = 2 tiles, K = 2 chunks.
// ---------------------------------------------------------------------------
__global__ __launch_bounds__(256) void embed_kernel(const float* __restrict__ x,
                                                    const float* __restrict__ Wemb,
                                                    const float* __restrict__ bemb,
                                                    f16* __restrict__ ep)
{
    __shared__ f16   Ws[64 * 32];     // W_emb padded [64][32]
    __shared__ float bs[32];

    int tid = threadIdx.x;
    for (int i = tid; i < 64 * 32; i += 256) {
        int k = i >> 5, c = i & 31;
        Ws[i] = (c < D_EMB) ? (f16)Wemb[k * D_EMB + c] : (f16)0.0f;
    }
    if (tid < 32) bs[tid] = (tid < D_EMB) ? bemb[tid] : 0.0f;
    __syncthreads();

    const int wave  = tid >> 5;
    const int lane  = tid & 31;
    const int khalf = lane >> 4;
    const int lm    = lane & 15;
    const long base = (long)blockIdx.x * 128 + (long)wave * 16;
    const long arow = base + lm;

    // A fragments: 16-bit A 16x32 layout -> two contiguous 8-element runs per lane
    v16h A[2];
#pragma unroll
    for (int kc = 0; kc < 2; ++kc) {
        const float* p0 = x + arow * D_IN + kc * 32 + khalf * 8;
        const float* p1 = p0 + 16;
        v8h lo, hi;
#pragma unroll
        for (int j = 0; j < 8; ++j) { lo[j] = (f16)p0[j]; hi[j] = (f16)p1[j]; }
        A[kc] = __builtin_shufflevector(lo, hi, 0,1,2,3,4,5,6,7,8,9,10,11,12,13,14,15);
    }

#pragma unroll
    for (int nt = 0; nt < 2; ++nt) {
        int col = nt * 16 + lm;
        float bb = bs[col];
        v8f acc;
#pragma unroll
        for (int r = 0; r < 8; ++r) acc[r] = bb;
#pragma unroll
        for (int kc = 0; kc < 2; ++kc) {
            v16h Bf;   // 16-bit B 32x16: lane = col, lane-half h = K 8h..8h+7 / 16+8h..16+8h+7
#pragma unroll
            for (int j = 0; j < 8; ++j) {
                int k = kc * 32 + khalf * 8 + 2 * (j & 3) + (j >> 2) * 16;
                Bf[2 * j]     = Ws[k * 32 + col];
                Bf[2 * j + 1] = Ws[(k + 1) * 32 + col];
            }
            acc = __builtin_amdgcn_wmma_f32_16x16x32_f16(false, A[kc], false, Bf,
                                                         (short)0, acc, false, false);
        }
#pragma unroll
        for (int r = 0; r < 8; ++r) {
            long row = base + r + 8 * khalf;       // D row: M = r + 8*khalf
            int  t   = (int)(row & (T_SZ - 1));
            int  b   = (int)(row >> 12);
            float v  = (col < D_EMB) ? sigmoidf_(acc[r]) : 0.0f;  // pad cols must be 0
            ep[(long)t * (B_SZ * 32) + b * 32 + col] = (f16)v;
        }
    }
}

// ---------------------------------------------------------------------------
// K3: persistent LSTM scan. 1 block, 1024 threads = 32 waves on one WGP.
//     Wave w owns gate-column strip [16w, 16w+16); gate id = w/8.
//     B fragments (5 K-chunks, 40 VGPRs) stay resident across all 4096 steps.
// LDS: z f16[64][160] @0 (20480) | gates f32[64][512] @20480 (131072)
//      | c  f32[64][128] @151552 (32768)  -> 184320 bytes (< 320KB WGP LDS)
// ---------------------------------------------------------------------------
#define LDS_Z      0
#define LDS_GATES  20480
#define LDS_C      151552
#define LDS_TOTAL  184320

__global__ __launch_bounds__(1024, 1) void lstm_kernel(const f16*  __restrict__ ep,
                                                       const f16*  __restrict__ Wz,
                                                       const float* __restrict__ bz,
                                                       const float* __restrict__ Wout,
                                                       const float* __restrict__ bout,
                                                       float* __restrict__ out)
{
    extern __shared__ char smem[];
    f16*   z     = (f16*)(smem + LDS_Z);       // [64][160]
    float* gates = (float*)(smem + LDS_GATES); // [64][512]
    float* cst   = (float*)(smem + LDS_C);     // [64][128]

    const int tid   = threadIdx.x;
    const int wave  = tid >> 5;
    const int lane  = tid & 31;
    const int khalf = lane >> 4;
    const int lm    = lane & 15;
    const int col   = wave * 16 + lm;     // global gate column 0..511
    const int gate  = wave >> 3;          // 0=f 1=i 2=g(tanh) 3=o

    // ---- init state: c = 0, h-part of z = 0
    for (int i = tid; i < B_SZ * H_SZ; i += 1024) {
        cst[i] = 0.0f;
        z[(i >> 7) * KZ + 32 + (i & 127)] = (f16)0.0f;
    }

    // ---- resident B fragments: Wz[k][col], K-chunk kc covers k in [32kc, 32kc+32)
    v16h Bf[5];
#pragma unroll
    for (int kc = 0; kc < 5; ++kc) {
#pragma unroll
        for (int j = 0; j < 8; ++j) {
            int k = kc * 32 + khalf * 8 + 2 * (j & 3) + (j >> 2) * 16;
            Bf[kc][2 * j]     = Wz[(long)k * NG + col];
            Bf[kc][2 * j + 1] = Wz[(long)(k + 1) * NG + col];
        }
    }
    const float bb = bz[col];
    __syncthreads();

    // per-thread async e_t -> LDS routing: thread copies one b32 word
    const unsigned zlds = (unsigned)(size_t)(void*)z;          // LDS offset of z
    const int      eb   = tid >> 4;                            // batch row 0..63
    const int      ew   = tid & 15;                            // word in 64B row
    const unsigned ldsa = zlds + (unsigned)(eb * (KZ * 2) + ew * 4);

    for (int t = 0; t < T_SZ; ++t) {
        // ---- stage e_t (4KB) straight into z[:,0:32] via async tensor path
        {
            unsigned long long ga =
                (unsigned long long)(size_t)((const char*)ep + (size_t)t * 4096 + (size_t)tid * 4);
            asm volatile("global_load_async_to_lds_b32 %0, %1, off"
                         :: "v"(ldsa), "v"(ga) : "memory");
            asm volatile("s_wait_asynccnt 0x0" ::: "memory");
        }
        __syncthreads();

        // ---- gate GEMM: gates_pre[64,512] = z[64,160] x Wz[160,512] + bias
#pragma unroll
        for (int mt = 0; mt < 4; ++mt) {
            const int arow = mt * 16 + lm;
            v8f acc;
#pragma unroll
            for (int r = 0; r < 8; ++r) acc[r] = bb;
#pragma unroll
            for (int kc = 0; kc < 5; ++kc) {
                const v8h* plo = (const v8h*)(z + arow * KZ + kc * 32 + khalf * 8);
                const v8h* phi = (const v8h*)(z + arow * KZ + kc * 32 + 16 + khalf * 8);
                v16h A = __builtin_shufflevector(*plo, *phi,
                                                 0,1,2,3,4,5,6,7,8,9,10,11,12,13,14,15);
                acc = __builtin_amdgcn_wmma_f32_16x16x32_f16(false, A, false, Bf[kc],
                                                             (short)0, acc, false, false);
            }
#pragma unroll
            for (int r = 0; r < 8; ++r) {
                int brow = mt * 16 + r + 8 * khalf;
                float v  = acc[r];
                float a  = (gate == 2) ? tanhf_(v) : sigmoidf_(v);
                gates[brow * NG + col] = a;
            }
        }
        __syncthreads();

        // ---- elementwise state update: 8192 elems / 1024 threads
#pragma unroll
        for (int e = 0; e < 8; ++e) {
            int idx = tid + e * 1024;
            int b = idx >> 7, j = idx & 127;
            float f = gates[b * NG + j];
            float i = gates[b * NG + 128 + j];
            float g = gates[b * NG + 256 + j];
            float o = gates[b * NG + 384 + j];
            float c = cst[b * H_SZ + j] * f + i * g;
            cst[b * H_SZ + j] = c;
            z[b * KZ + 32 + j] = (f16)(tanhf_(c) * o);   // h for next step, already f16
        }
        __syncthreads();
    }

    // ---- output projection: out[b] = sigmoid(h[b] . Wout + bout)
    if (tid < B_SZ) {
        float s = bout[0];
#pragma unroll 4
        for (int j = 0; j < H_SZ; ++j)
            s += (float)z[tid * KZ + 32 + j] * Wout[j];
        out[tid] = sigmoidf_(s);
    }
}

// ---------------------------------------------------------------------------
extern "C" void kernel_launch(void* const* d_in, const int* in_sizes, int n_in,
                              void* d_out, int out_size, void* d_ws, size_t ws_size,
                              hipStream_t stream)
{
    (void)in_sizes; (void)n_in; (void)out_size; (void)ws_size;

    const float* x    = (const float*)d_in[0];
    const float* Wemb = (const float*)d_in[1];
    const float* bemb = (const float*)d_in[2];
    const float* Wf   = (const float*)d_in[3];
    const float* bf   = (const float*)d_in[4];
    const float* Wi   = (const float*)d_in[5];
    const float* bi   = (const float*)d_in[6];
    const float* Wc   = (const float*)d_in[7];
    const float* bc   = (const float*)d_in[8];
    const float* Wo   = (const float*)d_in[9];
    const float* bo   = (const float*)d_in[10];
    const float* Wout = (const float*)d_in[11];
    const float* bout = (const float*)d_in[12];

    // workspace layout
    char*  ws = (char*)d_ws;
    f16*   Wz = (f16*)ws;                        // 160*512*2 = 163840 B
    float* bz = (float*)(ws + 163840);           // 512*4    = 2048 B -> ends 165888 (256-aligned)
    f16*   ep = (f16*)(ws + 165888);             // 4096*64*32*2 = 16 MB

    prep_wz_kernel<<<(KZ * NG + 255) / 256, 256, 0, stream>>>(Wf, bf, Wi, bi, Wc, bc, Wo, bo, Wz, bz);

    // 262144 rows / (16 rows * 8 waves) = 2048 workgroups
    embed_kernel<<<2048, 256, 0, stream>>>(x, Wemb, bemb, ep);

    (void)hipFuncSetAttribute((const void*)lstm_kernel,
                              hipFuncAttributeMaxDynamicSharedMemorySize, LDS_TOTAL);
    lstm_kernel<<<1, 1024, LDS_TOTAL, stream>>>(ep, Wz, bz, Wout, bout, (float*)d_out);
}
